// MaskedLinear_18562848653614
// MI455X (gfx1250) — matrix operand order repaired and checked by reading, not compiled, not run on hardware
//
#include <hip/hip_runtime.h>

typedef __attribute__((ext_vector_type(16))) __bf16 v16bf;
typedef __attribute__((ext_vector_type(8)))  __bf16 v8bf;
typedef __attribute__((ext_vector_type(8)))  float  v8f;

#define IN_F   4096
#define OUT_F  4096
#define BATCH  512
#define NNZ    1600000
#define WELEMS (OUT_F * IN_F)   // 16,777,216

#define KSTAGE  64                    // K elements staged per LDS stage
#define NSTAGES (IN_F / KSTAGE)       // 64 stages

// ---------------------------------------------------------------------------
// CDNA5 async DMA: global -> LDS, 16 bytes per lane, tracked by ASYNCcnt.
// (CDNA5 ISA; inline asm is portable across toolchains)
// ---------------------------------------------------------------------------
__device__ __forceinline__ void async_copy_b128(const __bf16* gptr,
                                                const __bf16* lptr) {
    // Low 32 bits of a flat LDS pointer are the workgroup-relative LDS offset
    // (flat->LDS mapping truncates to addr[31:0]; async op adds LDS_BASE).
    unsigned lds_off = (unsigned)(unsigned long long)lptr;
    asm volatile("global_load_async_to_lds_b128 %0, %1, off"
                 :: "v"(lds_off), "v"(gptr)
                 : "memory");
}

__device__ __forceinline__ void wait_async_le4() {
    asm volatile("s_wait_asynccnt 0x4" ::: "memory");
}
__device__ __forceinline__ void wait_async_le0() {
    asm volatile("s_wait_asynccnt 0x0" ::: "memory");
}

// ---------------------------------------------------------------------------
// Phase 1a: winner[p] = -1 (vectorized int4 stores)
// ---------------------------------------------------------------------------
__global__ void ml_init_winner(int4* __restrict__ winner4) {
    int idx = blockIdx.x * blockDim.x + threadIdx.x;   // WELEMS/4 threads exactly
    winner4[idx] = make_int4(-1, -1, -1, -1);
}

// ---------------------------------------------------------------------------
// Phase 1b: deterministic last-write-wins scatter via atomicMax of nnz index
// ---------------------------------------------------------------------------
__global__ void ml_scatter_winner(int* __restrict__ winner,
                                  const int* __restrict__ row_idx,
                                  const int* __restrict__ col_idx,
                                  int nnz) {
    int i = blockIdx.x * blockDim.x + threadIdx.x;
    if (i < nnz) {
        int r = row_idx[i];
        int c = col_idx[i];
        atomicMax(&winner[r * IN_F + c], i);
    }
}

// ---------------------------------------------------------------------------
// Phase 1c: materialize W as bf16 hi/lo split (row-major [out_f][in_f])
// ---------------------------------------------------------------------------
__global__ void ml_build_w(const int* __restrict__ winner,
                           const float* __restrict__ w1d,
                           __bf16* __restrict__ whi,
                           __bf16* __restrict__ wlo) {
    int p = blockIdx.x * blockDim.x + threadIdx.x;     // WELEMS threads exactly
    int win = winner[p];
    float w = (win >= 0) ? w1d[win] : 0.0f;
    __bf16 h = (__bf16)w;
    float  l = w - (float)h;
    whi[p] = h;
    wlo[p] = (__bf16)l;
}

// ---------------------------------------------------------------------------
// Phase 1d: split x into bf16 hi/lo
// ---------------------------------------------------------------------------
__global__ void ml_split_x(const float* __restrict__ x,
                           __bf16* __restrict__ xhi,
                           __bf16* __restrict__ xlo) {
    int p = blockIdx.x * blockDim.x + threadIdx.x;     // BATCH*IN_F threads exactly
    float v = x[p];
    __bf16 h = (__bf16)v;
    float  l = v - (float)h;
    xhi[p] = h;
    xlo[p] = (__bf16)l;
}

// ---------------------------------------------------------------------------
// Phase 2: out = xhi*Whi^T + xhi*Wlo^T + xlo*Whi^T + bias  (bf16x3 ~ f32)
//
// Block = 256 threads = 8 waves. Wave w owns M-tile (blockIdx.y*128 + w*16);
// all 8 waves share the same 4 N-tiles at blockIdx.x*64, so the B tile
// (W rows, hi+lo) is staged ONCE per block in LDS via async DMA,
// triple-buffered, and consumed with ds_load_b128.
//
// LDS: shB[3][2][64][KSTAGE] bf16 = 3 * 16KB = 48KB.
//
// Per-ISA CDNA5 WMMA bf16 layouts (wave32):
//   A 16x32: lane L -> row M=L%16; K chunks [h*8,+8) and [16+h*8,+8)
//   B 32x16: lane L -> col N=L%16; K chunks [h*8,+8) and [16+h*8,+8)
//   C 16x16: v8f elem j -> M = j + h*8, N = L%16        (h = L/16)
// ---------------------------------------------------------------------------
__global__ __launch_bounds__(256) void ml_gemm_bf16x3(
        const __bf16* __restrict__ Whi, const __bf16* __restrict__ Wlo,
        const __bf16* __restrict__ Xhi, const __bf16* __restrict__ Xlo,
        const float* __restrict__ bias, float* __restrict__ out) {

    __shared__ __bf16 shB[3][2][64][KSTAGE];   // [buf][hi/lo][n_row][k]

    const int tid  = threadIdx.x;
    const int lane = tid & 31;
    const int wave = tid >> 5;
    const int half = lane >> 4;     // 0 or 1
    const int l16  = lane & 15;

    const int m_base = blockIdx.y * 128 + wave * 16;
    const int n_base = blockIdx.x * 64;

    const __bf16* xh_row = Xhi + (size_t)(m_base + l16) * IN_F;
    const __bf16* xl_row = Xlo + (size_t)(m_base + l16) * IN_F;

    // Per-thread B-staging assignment: 2 x 16B chunks per matrix per stage.
    // 64 rows * KSTAGE(64) * 2B = 8KB per matrix = 512 chunks; 256 threads.
    const int c0   = tid;            // chunk ids: tid, tid+256
    const int c1   = tid + 256;
    const int r0   = c0 >> 3,  e0 = (c0 & 7) * 8;   // row, elem offset
    const int r1   = c1 >> 3,  e1 = (c1 & 7) * 8;
    const __bf16* gh0 = Whi + (size_t)(n_base + r0) * IN_F + e0;
    const __bf16* gh1 = Whi + (size_t)(n_base + r1) * IN_F + e1;
    const __bf16* gl0 = Wlo + (size_t)(n_base + r0) * IN_F + e0;
    const __bf16* gl1 = Wlo + (size_t)(n_base + r1) * IN_F + e1;

    auto issue_stage = [&](int k0, int buf) {
        async_copy_b128(gh0 + k0, &shB[buf][0][r0][e0]);
        async_copy_b128(gh1 + k0, &shB[buf][0][r1][e1]);
        async_copy_b128(gl0 + k0, &shB[buf][1][r0][e0]);
        async_copy_b128(gl1 + k0, &shB[buf][1][r1][e1]);
    };

    // Prologue: stage 0 in flight.
    issue_stage(0, 0);

    v8f acc[4] = {};
    const int aoff = half * 8;      // A chunk0 K-offset within 32-wide window
    const int boff = half * 8;      // B chunk0 K-offset within 32-wide window

    int buf = 0;
    for (int s = 0; s < NSTAGES; ++s) {
        const int k0   = s * KSTAGE;
        const int nbuf = (buf == 2) ? 0 : buf + 1;

        if (s + 1 < NSTAGES) {
            issue_stage(k0 + KSTAGE, nbuf);
            // Prefetch next stage's A rows into L2 while DMA runs.
            __builtin_prefetch(xh_row + k0 + KSTAGE, 0, 1);
            __builtin_prefetch(xl_row + k0 + KSTAGE, 0, 1);
            wait_async_le4();       // 4 newer (next-stage) may remain in flight
        } else {
            wait_async_le0();
        }
        __syncthreads();            // stage s visible to all 8 waves

#pragma unroll
        for (int ks = 0; ks < KSTAGE / 32; ++ks) {
            const int k = k0 + ks * 32;

            // A fragments (hi and lo) from global: two 16B chunks each.
            v8bf a0h = *(const v8bf*)(xh_row + k + aoff);
            v8bf a1h = *(const v8bf*)(xh_row + k + aoff + 16);
            v8bf a0l = *(const v8bf*)(xl_row + k + aoff);
            v8bf a1l = *(const v8bf*)(xl_row + k + aoff + 16);
            v16bf ah = __builtin_shufflevector(a0h, a1h,
                            0,1,2,3,4,5,6,7,8,9,10,11,12,13,14,15);
            v16bf al = __builtin_shufflevector(a0l, a1l,
                            0,1,2,3,4,5,6,7,8,9,10,11,12,13,14,15);

#pragma unroll
            for (int t = 0; t < 4; ++t) {
                const __bf16* bh_p = &shB[buf][0][t * 16 + l16][ks * 32 + boff];
                const __bf16* bl_p = &shB[buf][1][t * 16 + l16][ks * 32 + boff];
                v8bf b0h = *(const v8bf*)(bh_p);
                v8bf b1h = *(const v8bf*)(bh_p + 16);
                v8bf b0l = *(const v8bf*)(bl_p);
                v8bf b1l = *(const v8bf*)(bl_p + 16);
                v16bf bh = __builtin_shufflevector(b0h, b1h,
                                0,1,2,3,4,5,6,7,8,9,10,11,12,13,14,15);
                v16bf bl = __builtin_shufflevector(b0l, b1l,
                                0,1,2,3,4,5,6,7,8,9,10,11,12,13,14,15);

                // acc += ah*bh + ah*bl + al*bh   (bf16x3 f32 emulation)
                acc[t] = __builtin_amdgcn_wmma_f32_16x16x32_bf16(
                             false, ah, false, bh, (short)0, acc[t], false, false);
                acc[t] = __builtin_amdgcn_wmma_f32_16x16x32_bf16(
                             false, ah, false, bl, (short)0, acc[t], false, false);
                acc[t] = __builtin_amdgcn_wmma_f32_16x16x32_bf16(
                             false, al, false, bh, (short)0, acc[t], false, false);
            }
        }
        buf = nbuf;
    }

    // Epilogue: D elem j -> (m_base + half*8 + j, n_tile + l16); add bias.
    const int mrow = m_base + half * 8;
#pragma unroll
    for (int t = 0; t < 4; ++t) {
        int n = n_base + t * 16 + l16;
        float b = bias[n];
#pragma unroll
        for (int j = 0; j < 8; ++j) {
            out[(size_t)(mrow + j) * OUT_F + n] = acc[t][j] + b;
        }
    }
}

// ---------------------------------------------------------------------------
extern "C" void kernel_launch(void* const* d_in, const int* in_sizes, int n_in,
                              void* d_out, int out_size, void* d_ws, size_t ws_size,
                              hipStream_t stream) {
    const float* x    = (const float*)d_in[0];
    const float* w1d  = (const float*)d_in[1];
    const float* bias = (const float*)d_in[2];
    const int*   rowi = (const int*)d_in[3];
    const int*   coli = (const int*)d_in[4];

    char* ws = (char*)d_ws;
    int*    winner = (int*)ws;                               //  64 MB
    __bf16* Whi    = (__bf16*)(ws + ((size_t)64  << 20));    //  32 MB
    __bf16* Wlo    = (__bf16*)(ws + ((size_t)96  << 20));    //  32 MB
    __bf16* Xhi    = (__bf16*)(ws + ((size_t)128 << 20));    //   4 MB
    __bf16* Xlo    = (__bf16*)(ws + ((size_t)132 << 20));    //   4 MB

    // Phase 1: deterministic scatter + bf16 hi/lo materialization
    ml_init_winner<<<(WELEMS / 4) / 256, 256, 0, stream>>>((int4*)winner);
    ml_scatter_winner<<<(NNZ + 255) / 256, 256, 0, stream>>>(winner, rowi, coli, NNZ);
    ml_build_w<<<WELEMS / 256, 256, 0, stream>>>(winner, w1d, Whi, Wlo);
    ml_split_x<<<(BATCH * IN_F) / 256, 256, 0, stream>>>(x, Xhi, Xlo);

    // Phase 2: WMMA GEMM, async-LDS-staged B, bias epilogue
    dim3 grid(OUT_F / 64, BATCH / 128);   // (64, 4)
    ml_gemm_bf16x3<<<grid, 256, 0, stream>>>(Whi, Wlo, Xhi, Xlo, bias, (float*)d_out);
}
